// MPNN_10342281248775
// MI455X (gfx1250) — matrix operand order repaired and checked by reading, not compiled, not run on hardware
//
#include <hip/hip_runtime.h>
#include <stdint.h>

// ---------- WMMA plumbing (gfx1250, wave32) ----------
typedef __attribute__((ext_vector_type(16))) __bf16 v16bf;
typedef __attribute__((ext_vector_type(8)))  float  v8f;

union ABits { uint32_t u[8]; v16bf v; };

__device__ __forceinline__ uint32_t f2bf1(float f) {
  uint32_t u = __float_as_uint(f);
  return (u + 0x7FFFu + ((u >> 16) & 1u)) >> 16;   // round-to-nearest-even bf16
}
__device__ __forceinline__ uint32_t pk2bf(float lo, float hi) {
  return f2bf1(lo) | (f2bf1(hi) << 16);
}

__device__ __forceinline__ v8f wmma_bf16(v16bf a, v16bf b, v8f c) {
  // (neg_a, A, neg_b, B, c_mod, C, reuse_a, reuse_b)
  return __builtin_amdgcn_wmma_f32_16x16x32_bf16(false, a, false, b, (short)0, c,
                                                 false, false);
}

// A operand, 16-bit A 16x32 layout (ISA 7.12.2): per lane two contiguous
// 8-elem chunks at [kbase + hs*8] and [kbase + 16 + hs*8].
__device__ __forceinline__ v16bf load_matA_bf(const uint16_t* row, int kbase, int hs) {
  ABits t;
  const uint4* p0 = (const uint4*)(row + kbase + hs * 8);
  const uint4* p1 = (const uint4*)(row + kbase + 16 + hs * 8);
  uint4 a = *p0, b = *p1;
  t.u[0] = a.x; t.u[1] = a.y; t.u[2] = a.z; t.u[3] = a.w;
  t.u[4] = b.x; t.u[5] = b.y; t.u[6] = b.z; t.u[7] = b.w;
  return t.v;
}

// Same A layout but source is fp32 in global memory (convert on the fly).
__device__ __forceinline__ v16bf load_matA_f32(const float* row, int kbase, int hs) {
  ABits t;
  const float4* p0 = (const float4*)(row + kbase + hs * 8);
  const float4* p1 = (const float4*)(row + kbase + 16 + hs * 8);
  float4 a0 = p0[0], a1 = p0[1];
  float4 b0 = p1[0], b1 = p1[1];
  t.u[0] = pk2bf(a0.x, a0.y); t.u[1] = pk2bf(a0.z, a0.w);
  t.u[2] = pk2bf(a1.x, a1.y); t.u[3] = pk2bf(a1.z, a1.w);
  t.u[4] = pk2bf(b0.x, b0.y); t.u[5] = pk2bf(b0.z, b0.w);
  t.u[6] = pk2bf(b1.x, b1.y); t.u[7] = pk2bf(b1.z, b1.w);
  return t.v;
}

// B operand, 16-bit B 32x16 layout: lane holds column n = lane&15 of B, i.e.
// row n of W^T; two 8-elem K chunks at [kbase + hs*8] and [kbase + 16 + hs*8].
__device__ __forceinline__ v16bf load_matB_bf(const uint16_t* row, int kbase, int hs) {
  ABits t;
  const uint4* p0 = (const uint4*)(row + kbase + hs * 8);
  const uint4* p1 = (const uint4*)(row + kbase + 16 + hs * 8);
  uint4 a = *p0, b = *p1;
  t.u[0] = a.x; t.u[1] = a.y; t.u[2] = a.z; t.u[3] = a.w;
  t.u[4] = b.x; t.u[5] = b.y; t.u[6] = b.z; t.u[7] = b.w;
  return t.v;
}

__device__ __forceinline__ float elu1(float z) {
  return z > 0.f ? z : (expf(z) - 1.f);
}

// ---------- kernel 1: per-node embed + F MLP; also emit e as bf16 ----------
__global__ void __launch_bounds__(128) node_kernel(
    const float* __restrict__ x, const float* __restrict__ theta,
    const float* __restrict__ emb_W, const float* __restrict__ emb_b,
    const float* __restrict__ F_W1, const float* __restrict__ F_b1,
    const float* __restrict__ F_W2, const float* __restrict__ F_b2,
    uint16_t* __restrict__ e_bf, float* __restrict__ fout, int N) {
  __shared__ __align__(16) uint16_t Wt1[64 * 128];   // F_W1^T, bf16
  __shared__ __align__(16) uint16_t Wt2[64 * 64];    // F_W2^T, bf16
  __shared__ float b1s[64], b2s[64];
  __shared__ __align__(16) uint16_t hid[4][16 * 64]; // per-wave hidden staging

  const int tid = threadIdx.x;
  for (int i = tid; i < 128 * 64; i += 128) {
    int k = i >> 6, n = i & 63;
    Wt1[n * 128 + k] = (uint16_t)f2bf1(F_W1[i]);
  }
  for (int i = tid; i < 64 * 64; i += 128) {
    int k = i >> 6, n = i & 63;
    Wt2[n * 64 + k] = (uint16_t)f2bf1(F_W2[i]);
  }
  if (tid < 64) { b1s[tid] = F_b1[tid]; b2s[tid] = F_b2[tid]; }
  __syncthreads();

  const int wave = tid >> 5, lane = tid & 31;
  const int lrow = lane & 15, hs = lane >> 4;
  uint16_t* H = hid[wave];
  const int numTiles = N >> 4;

  for (int tile = blockIdx.x * 4 + wave; tile < numTiles; tile += gridDim.x * 4) {
    const int node = tile * 16 + lrow;
    const float xv = x[node];
    v8f acc[4];
#pragma unroll
    for (int t = 0; t < 4; ++t)
#pragma unroll
      for (int v = 0; v < 8; ++v) acc[t][v] = 0.f;

    // ---- layer 1: K = 128 = [e(64) | theta(64)], 4 K-steps of 32
#pragma unroll
    for (int j = 0; j < 4; ++j) {
      v16bf a;
      if (j < 2) {
        // compute e chunks directly in A layout and persist them as bf16
        ABits t;
        const int kc0 = j * 32 + hs * 8;
        const int kc1 = j * 32 + 16 + hs * 8;
        float ch[16];
#pragma unroll
        for (int q = 0; q < 8; ++q) ch[q]     = fmaf(xv, emb_W[kc0 + q], emb_b[kc0 + q]);
#pragma unroll
        for (int q = 0; q < 8; ++q) ch[8 + q] = fmaf(xv, emb_W[kc1 + q], emb_b[kc1 + q]);
#pragma unroll
        for (int q = 0; q < 8; ++q) t.u[q] = pk2bf(ch[2 * q], ch[2 * q + 1]);
        uint16_t* erow = e_bf + (size_t)node * 64;
        *(uint4*)(erow + kc0) = make_uint4(t.u[0], t.u[1], t.u[2], t.u[3]);
        *(uint4*)(erow + kc1) = make_uint4(t.u[4], t.u[5], t.u[6], t.u[7]);
        a = t.v;
      } else {
        a = load_matA_f32(theta + (size_t)node * 64, (j - 2) * 32, hs);
      }
#pragma unroll
      for (int tt = 0; tt < 4; ++tt) {
        v16bf b = load_matB_bf(Wt1 + (tt * 16 + lrow) * 128, j * 32, hs);
        acc[tt] = wmma_bf16(a, b, acc[tt]);
      }
    }

    // ---- bias + elu, restage hidden (C layout -> row-major LDS)
#pragma unroll
    for (int tt = 0; tt < 4; ++tt) {
      const float bb = b1s[tt * 16 + lrow];
#pragma unroll
      for (int v = 0; v < 8; ++v) {
        const int m = v + hs * 8;                      // C row for this lane
        H[m * 64 + tt * 16 + lrow] = (uint16_t)f2bf1(elu1(acc[tt][v] + bb));
      }
    }

    // ---- layer 2: K = 64, 2 K-steps
    v8f acc2[4];
#pragma unroll
    for (int t = 0; t < 4; ++t)
#pragma unroll
      for (int v = 0; v < 8; ++v) acc2[t][v] = 0.f;
#pragma unroll
    for (int j = 0; j < 2; ++j) {
      v16bf a = load_matA_bf(H + lrow * 64, j * 32, hs);
#pragma unroll
      for (int tt = 0; tt < 4; ++tt) {
        v16bf b = load_matB_bf(Wt2 + (tt * 16 + lrow) * 64, j * 32, hs);
        acc2[tt] = wmma_bf16(a, b, acc2[tt]);
      }
    }
#pragma unroll
    for (int tt = 0; tt < 4; ++tt) {
      const float bb = b2s[tt * 16 + lrow];
#pragma unroll
      for (int v = 0; v < 8; ++v) {
        const int m = v + hs * 8;
        fout[(size_t)(tile * 16 + m) * 64 + tt * 16 + lrow] = acc2[tt][v] + bb;
      }
    }
  }
}

// ---------- kernel 2: per-edge G MLP + atomic scatter-add ----------
__global__ void __launch_bounds__(128) edge_kernel(
    const long long* __restrict__ edge_index, const float* __restrict__ weight,
    const float* __restrict__ G_W1, const float* __restrict__ G_b1,
    const float* __restrict__ G_W2, const float* __restrict__ G_b2,
    const uint16_t* __restrict__ e_bf, float* __restrict__ agg, int E) {
  __shared__ __align__(16) uint16_t Wt1[64 * 192];   // G_W1^T, bf16
  __shared__ __align__(16) uint16_t Wt2[64 * 64];    // G_W2^T, bf16
  __shared__ float b1s[64], b2s[64];
  __shared__ __align__(16) uint16_t hid[4][16 * 64];

  const int tid = threadIdx.x;
  for (int i = tid; i < 192 * 64; i += 128) {
    int k = i >> 6, n = i & 63;
    Wt1[n * 192 + k] = (uint16_t)f2bf1(G_W1[i]);
  }
  for (int i = tid; i < 64 * 64; i += 128) {
    int k = i >> 6, n = i & 63;
    Wt2[n * 64 + k] = (uint16_t)f2bf1(G_W2[i]);
  }
  if (tid < 64) { b1s[tid] = G_b1[tid]; b2s[tid] = G_b2[tid]; }
  __syncthreads();

  const int wave = tid >> 5, lane = tid & 31;
  const int lrow = lane & 15, hs = lane >> 4;
  uint16_t* H = hid[wave];
  const int numTiles = E >> 4;

  for (int tile = blockIdx.x * 4 + wave; tile < numTiles; tile += gridDim.x * 4) {
    const int e0 = tile * 16;
    const int srcI = (int)edge_index[e0 + lrow];
    const int dstI = (int)edge_index[(size_t)E + e0 + lrow];

    const uint16_t* rowXi = e_bf + (size_t)dstI * 64;
    const uint16_t* rowXj = e_bf + (size_t)srcI * 64;
    const float*    rowW  = weight + (size_t)(e0 + lrow) * 64;

    v8f acc[4];
#pragma unroll
    for (int t = 0; t < 4; ++t)
#pragma unroll
      for (int v = 0; v < 8; ++v) acc[t][v] = 0.f;

    // ---- layer 1: K = 192 = [x_i(64) | x_j(64) | w(64)], 6 K-steps of 32
#pragma unroll
    for (int j = 0; j < 6; ++j) {
      v16bf a;
      if (j < 2)      a = load_matA_bf(rowXi, j * 32, hs);
      else if (j < 4) a = load_matA_bf(rowXj, (j - 2) * 32, hs);
      else            a = load_matA_f32(rowW, (j - 4) * 32, hs);
#pragma unroll
      for (int tt = 0; tt < 4; ++tt) {
        v16bf b = load_matB_bf(Wt1 + (tt * 16 + lrow) * 192, j * 32, hs);
        acc[tt] = wmma_bf16(a, b, acc[tt]);
      }
    }

    // ---- bias + elu, restage hidden
#pragma unroll
    for (int tt = 0; tt < 4; ++tt) {
      const float bb = b1s[tt * 16 + lrow];
#pragma unroll
      for (int v = 0; v < 8; ++v) {
        const int m = v + hs * 8;
        H[m * 64 + tt * 16 + lrow] = (uint16_t)f2bf1(elu1(acc[tt][v] + bb));
      }
    }

    // ---- layer 2: K = 64
    v8f acc2[4];
#pragma unroll
    for (int t = 0; t < 4; ++t)
#pragma unroll
      for (int v = 0; v < 8; ++v) acc2[t][v] = 0.f;
#pragma unroll
    for (int j = 0; j < 2; ++j) {
      v16bf a = load_matA_bf(H + lrow * 64, j * 32, hs);
#pragma unroll
      for (int tt = 0; tt < 4; ++tt) {
        v16bf b = load_matB_bf(Wt2 + (tt * 16 + lrow) * 64, j * 32, hs);
        acc2[tt] = wmma_bf16(a, b, acc2[tt]);
      }
    }

    // ---- scatter messages: agg[dst[m]] += m_row  (f32 atomics, L2-resident)
#pragma unroll
    for (int tt = 0; tt < 4; ++tt) {
      const float bb = b2s[tt * 16 + lrow];
#pragma unroll
      for (int v = 0; v < 8; ++v) {
        const int m = v + hs * 8;
        const int d = __shfl(dstI, m, 32);   // dst index of edge-row m
        atomicAdd(&agg[(size_t)d * 64 + tt * 16 + lrow], acc2[tt][v] + bb);
      }
    }
  }
}

// ---------- kernel 3: h = f + agg ; out = h @ out_W + out_b ----------
__global__ void out_kernel(const float* __restrict__ f, const float* __restrict__ agg,
                           const float* __restrict__ out_W, const float* __restrict__ out_b,
                           float* __restrict__ out, int N) {
  const int gwave = (blockIdx.x * blockDim.x + threadIdx.x) >> 5;
  const int lane = threadIdx.x & 31;
  const int nwaves = (gridDim.x * blockDim.x) >> 5;
  const float w0 = out_W[lane], w1 = out_W[32 + lane];
  const float bias = out_b[0];
  for (int n = gwave; n < N; n += nwaves) {
    const float* fr = f + (size_t)n * 64;
    const float* ar = agg + (size_t)n * 64;
    float s = (fr[lane] + ar[lane]) * w0 + (fr[32 + lane] + ar[32 + lane]) * w1;
#pragma unroll
    for (int off = 16; off > 0; off >>= 1) s += __shfl_down(s, off, 32);
    if (lane == 0) out[n] = s + bias;
  }
}

// ---------- launch ----------
extern "C" void kernel_launch(void* const* d_in, const int* in_sizes, int n_in,
                              void* d_out, int out_size, void* d_ws, size_t ws_size,
                              hipStream_t stream) {
  (void)n_in; (void)ws_size;
  const float*     x     = (const float*)d_in[0];
  const long long* eidx  = (const long long*)d_in[1];
  const float*     theta = (const float*)d_in[2];
  const float*     wgt   = (const float*)d_in[3];
  const float*     emb_W = (const float*)d_in[4];
  const float*     emb_b = (const float*)d_in[5];
  const float*     F_W1  = (const float*)d_in[6];
  const float*     F_b1  = (const float*)d_in[7];
  const float*     F_W2  = (const float*)d_in[8];
  const float*     F_b2  = (const float*)d_in[9];
  const float*     G_W1  = (const float*)d_in[10];
  const float*     G_b1  = (const float*)d_in[11];
  const float*     G_W2  = (const float*)d_in[12];
  const float*     G_b2  = (const float*)d_in[13];
  const float*     out_W = (const float*)d_in[14];
  const float*     out_b = (const float*)d_in[15];

  const int N = in_sizes[0];       // 50000
  const int E = in_sizes[1] / 2;   // 800000

  float*    agg  = (float*)d_ws;
  float*    fbuf = agg + (size_t)N * 64;
  uint16_t* e_bf = (uint16_t*)(fbuf + (size_t)N * 64);

  hipMemsetAsync(agg, 0, (size_t)N * 64 * sizeof(float), stream);

  const int nTilesN = N / 16;
  const int nTilesE = E / 16;
  int gridN = (nTilesN + 3) / 4; if (gridN > 1024) gridN = 1024;
  int gridE = (nTilesE + 3) / 4; if (gridE > 2048) gridE = 2048;

  node_kernel<<<gridN, 128, 0, stream>>>(x, theta, emb_W, emb_b,
                                         F_W1, F_b1, F_W2, F_b2,
                                         e_bf, fbuf, N);
  edge_kernel<<<gridE, 128, 0, stream>>>(eidx, wgt, G_W1, G_b1, G_W2, G_b2,
                                         e_bf, agg, E);
  out_kernel<<<256, 256, 0, stream>>>(fbuf, agg, out_W, out_b, (float*)d_out, N);
}
